// Attention_14877766713476
// MI455X (gfx1250) — matrix operand order repaired and checked by reading, not compiled, not run on hardware
//
#include <hip/hip_runtime.h>

// ---------------------------------------------------------------------------
// MI455X (gfx1250) fused attention block.
//   - all matmuls via v_wmma_f32_16x16x32_bf16 (bf16 in, f32 accumulate)
//   - projection GEMMs: 64x128 block tile, DOUBLE-BUFFERED LDS staging via
//     the CDNA5 async path (global_load_async_to_lds_b128, partial
//     s_wait_asynccnt 0x3 so the next tile stays in flight during compute),
//     8 waves each computing a 32x32 register tile (2x2 WMMA accumulators)
//   - flash attention: per-wave 16-query tile, online softmax, LDS repack of
//     P from C-layout to A-layout, 16 WMMAs per 32-key chunk
// Intermediates (~180MB) largely fit the 192MB L2 -> matrix-core bound.
// ---------------------------------------------------------------------------

#define DD 2048
#define BB 2
#define SS 2048
#define HH 16
#define HD 128
#define BS (BB * SS)

typedef __attribute__((ext_vector_type(16))) __bf16 v16bf;
typedef __attribute__((ext_vector_type(8)))  __bf16 v8bf;
typedef __attribute__((ext_vector_type(8)))  float  v8f;

static __device__ __forceinline__ __bf16 f2bf(float f) {
    unsigned u = __builtin_bit_cast(unsigned, f);
    u += 0x7FFFu + ((u >> 16) & 1u);                // round-to-nearest-even
    unsigned short s = (unsigned short)(u >> 16);
    return __builtin_bit_cast(__bf16, s);
}

// low 32 bits of a generic shared-memory pointer == LDS byte offset
// (ISA: LDS aperture maps addr[31:0] directly to the LDS allocation)
static __device__ __forceinline__ unsigned lds_off(const void* p) {
    return (unsigned)(unsigned long long)p;
}

// async copy of 16 bytes global -> LDS (per-lane addresses), ASYNCcnt-tracked
static __device__ __forceinline__ void async_b128(unsigned lds, const void* g) {
    asm volatile("global_load_async_to_lds_b128 %0, %1, off"
                 :: "v"(lds), "v"((unsigned long long)g) : "memory");
}
static __device__ __forceinline__ void wait_async0() {
    asm volatile("s_wait_asynccnt 0x0" ::: "memory");
}
// wait for the *current* buffer's 3 loads; leave the next buffer's 3 in flight
static __device__ __forceinline__ void wait_async3() {
    asm volatile("s_wait_asynccnt 0x3" ::: "memory");
}

// -------------------- conversion kernels -----------------------------------

__global__ __launch_bounds__(256) void cvt_bf16_kernel(const float* __restrict__ in,
                                                       __bf16* __restrict__ out) {
    int i = blockIdx.x * 256 + threadIdx.x;
    out[i] = f2bf(in[i]);
}

// W[k][n] (f32, row-major DxD) -> Wt[n][k] (bf16): WMMA B-operand friendly
__global__ __launch_bounds__(256) void cvt_wt_kernel(const float* __restrict__ W,
                                                     __bf16* __restrict__ Wt) {
    int i = blockIdx.x * 256 + threadIdx.x;   // output index: i = n*D + k
    int n = i / DD, k = i % DD;
    Wt[i] = f2bf(W[(size_t)k * DD + n]);
}

// -------------------- WMMA GEMM: Y = A(bf16) @ W + bias --------------------
// Block tile 64(M) x 128(N), K-step 32, double-buffered async LDS staging.
// mode 0: f32 row-major [BS][D]    (Q, K pre-RoPE; final output)
// mode 1: bf16 transposed-per-head (V -> Vt[B][H][HD][S])
__global__ __launch_bounds__(256) void gemm_kernel(
    const __bf16* __restrict__ A,      // [BS][D] bf16
    const __bf16* __restrict__ Wt,     // [D][D] bf16, Wt[n][k] = W[k][n]
    const float*  __restrict__ bias,   // [D]
    float*        __restrict__ outF,   // mode 0
    __bf16*       __restrict__ outVt,  // mode 1
    int mode)
{
    __shared__ __attribute__((aligned(32))) __bf16 ldsA[2][64][32];   // 2 x 4KB
    __shared__ __attribute__((aligned(32))) __bf16 ldsB[2][128][32];  // 2 x 8KB

    const int tid  = threadIdx.x;
    const int lane = tid & 31;
    const int wave = tid >> 5;
    const int wr   = wave >> 2;               // 0..1  (M direction)
    const int wc   = wave & 3;                // 0..3  (N direction)
    const int half = lane >> 4, l16 = lane & 15;
    const int koffA = half * 16;              // A-operand 16 contiguous K
    const int koffB = half * 16;              // B-operand 16 contiguous K

    const int blockM = blockIdx.x * 64;
    const int blockN = blockIdx.y * 128;

    // staging assignment: A = 64 rows x 4 b128 chunks; B = 128 rows x 4, x2
    const int arow = tid >> 2, achk = (tid & 3) * 8;
    const int bn   = tid >> 2, bk   = (tid & 3) * 8;

    unsigned la[2], lb0[2], lb1[2];
#pragma unroll
    for (int buf = 0; buf < 2; ++buf) {
        la[buf]  = lds_off(&ldsA[buf][arow][achk]);
        lb0[buf] = lds_off(&ldsB[buf][bn][bk]);
        lb1[buf] = lds_off(&ldsB[buf][bn + 64][bk]);
    }
    const __bf16* gA  = A  + (size_t)(blockM + arow) * DD + achk;
    const __bf16* gB0 = Wt + (size_t)(blockN + bn) * DD + bk;
    const __bf16* gB1 = Wt + (size_t)(blockN + bn + 64) * DD + bk;

    v8f c00 = {}, c01 = {}, c10 = {}, c11 = {};

    auto stage = [&](int buf, int k) {
        async_b128(la[buf],  gA  + k);
        async_b128(lb0[buf], gB0 + k);
        async_b128(lb1[buf], gB1 + k);
    };
    auto compute = [&](int buf) {
        v8bf alo0 = *(const v8bf*)&ldsA[buf][wr * 32 + l16][koffA];
        v8bf ahi0 = *(const v8bf*)&ldsA[buf][wr * 32 + l16][koffA + 8];
        v8bf alo1 = *(const v8bf*)&ldsA[buf][wr * 32 + 16 + l16][koffA];
        v8bf ahi1 = *(const v8bf*)&ldsA[buf][wr * 32 + 16 + l16][koffA + 8];
        v16bf a0, a1;
#pragma unroll
        for (int i = 0; i < 8; ++i) {
            a0[i] = alo0[i]; a0[8 + i] = ahi0[i];
            a1[i] = alo1[i]; a1[8 + i] = ahi1[i];
        }
        v16bf b0 = *(const v16bf*)&ldsB[buf][wc * 32 + l16][koffB];
        v16bf b1 = *(const v16bf*)&ldsB[buf][wc * 32 + 16 + l16][koffB];
        c00 = __builtin_amdgcn_wmma_f32_16x16x32_bf16(false, a0, false, b0, (short)0, c00, false, false);
        c01 = __builtin_amdgcn_wmma_f32_16x16x32_bf16(false, a0, false, b1, (short)0, c01, false, false);
        c10 = __builtin_amdgcn_wmma_f32_16x16x32_bf16(false, a1, false, b0, (short)0, c10, false, false);
        c11 = __builtin_amdgcn_wmma_f32_16x16x32_bf16(false, a1, false, b1, (short)0, c11, false, false);
    };

    const int NSTEP = DD / 32;                // 64 (even)
    stage(0, 0);                              // prologue
    for (int i = 0; i < NSTEP; i += 2) {
        // ---- even step: compute buf0, buf1's stage in flight during compute
        stage(1, (i + 1) * 32);               // i+1 <= 63 always valid
        wait_async3();                        // buf0 ready; buf1 still loading
        __syncthreads();
        compute(0);
        __syncthreads();                      // all reads of buf0 done
        // ---- odd step: compute buf1, restage buf0 (unless done)
        if (i + 2 < NSTEP) {
            stage(0, (i + 2) * 32);
            wait_async3();
        } else {
            wait_async0();
        }
        __syncthreads();
        compute(1);
        __syncthreads();
    }

    // ---- epilogue: bias + store, per 16x16 tile
    v8f* ctile[2][2] = { { &c00, &c01 }, { &c10, &c11 } };
#pragma unroll
    for (int mt = 0; mt < 2; ++mt) {
#pragma unroll
        for (int nt = 0; nt < 2; ++nt) {
            v8f& c = *ctile[mt][nt];
            const int col = blockN + wc * 32 + nt * 16 + l16;
            const float bval = bias[col];
            if (mode == 0) {
#pragma unroll
                for (int r = 0; r < 8; ++r) {
                    int row = blockM + wr * 32 + mt * 16 + r + 8 * half;
                    outF[(size_t)row * DD + col] = c[r] + bval;
                }
            } else {
#pragma unroll
                for (int r = 0; r < 8; ++r) {
                    int row = blockM + wr * 32 + mt * 16 + r + 8 * half;
                    int bb = row / SS, s = row % SS;
                    int h  = col / HD, d = col % HD;
                    outVt[(((size_t)bb * HH + h) * HD + d) * SS + s] = f2bf(c[r] + bval);
                }
            }
        }
    }
}

// -------------------- RoPE (full-D rotation, pre-head-split) ---------------
__global__ __launch_bounds__(256) void rope_kernel(const float* __restrict__ in,
                                                   const float* __restrict__ cosT,
                                                   const float* __restrict__ sinT,
                                                   __bf16* __restrict__ out) {
    int idx = blockIdx.x * 256 + threadIdx.x;       // over BS*D
    int r = idx / DD, i = idx % DD;
    int s = r % SS;
    int j = (i < DD / 2) ? i + DD / 2 : i - DD / 2;
    float sign = (i < DD / 2) ? -1.0f : 1.0f;
    float v = in[(size_t)r * DD + i] * cosT[(size_t)s * DD + i]
            + sign * in[(size_t)r * DD + j] * sinT[(size_t)s * DD + i];
    out[idx] = f2bf(v);
}

// -------------------- Flash attention (causal, online softmax) -------------
__global__ __launch_bounds__(256) void fattn_kernel(
    const __bf16* __restrict__ Q,   // [BS][D] bf16 (RoPE'd)
    const __bf16* __restrict__ K,   // [BS][D] bf16 (RoPE'd)
    const __bf16* __restrict__ Vt,  // [B][H][HD][S] bf16
    __bf16*       __restrict__ O)   // [BS][D] bf16 context
{
    __shared__ __attribute__((aligned(32))) __bf16 lds[8][16][32];

    const int lane = threadIdx.x & 31;
    const int wave = threadIdx.x >> 5;
    const int gw   = blockIdx.x * 8 + wave;
    const int QT   = SS / 16;                  // 128 query tiles per (b,h)
    const int b    = gw / (HH * QT);
    const int h    = (gw / QT) % HH;
    const int qt   = gw % QT;
    const int q0   = qt * 16;
    const int half = lane >> 4, l16 = lane & 15;
    const int koffA = half * 16;
    const int koffB = half * 16;

    const __bf16* qbase = Q + (size_t)(b * SS + q0 + l16) * DD + h * HD;
    const __bf16* kbase = K + (size_t)(b * SS) * DD + h * HD;
    const __bf16* vbase = Vt + ((size_t)b * HH + h) * HD * SS;

    v8f acc[8];
#pragma unroll
    for (int t = 0; t < 8; ++t) acc[t] = (v8f){};
    float rowmax[8], rowsum[8];
#pragma unroll
    for (int r = 0; r < 8; ++r) { rowmax[r] = -__builtin_inff(); rowsum[r] = 0.0f; }

    const float sc = 0.08838834764831845f;     // 1/sqrt(HD)

    for (int k0 = 0; k0 <= q0 + 15; k0 += 32) {
        // prefetch next chunk's K rows / V columns while this chunk computes
        if (k0 + 32 <= q0 + 15) {
            __builtin_prefetch(kbase + (size_t)(k0 + 32 + l16) * DD, 0, 1);
            __builtin_prefetch(vbase + (size_t)l16 * SS + k0 + 32, 0, 1);
        }
        // ---- scores: S0 = keys [k0..k0+15], S1 = keys [k0+16..k0+31]
        v8f s0 = {}, s1 = {};
#pragma unroll
        for (int dk = 0; dk < HD; dk += 32) {
            v8bf alo = *(const v8bf*)(qbase + dk + koffA);
            v8bf ahi = *(const v8bf*)(qbase + dk + koffA + 8);
            v16bf a;
#pragma unroll
            for (int i = 0; i < 8; ++i) { a[i] = alo[i]; a[8 + i] = ahi[i]; }
            v16bf b0 = *(const v16bf*)(kbase + (size_t)(k0 + l16) * DD + dk + koffB);
            v16bf b1 = *(const v16bf*)(kbase + (size_t)(k0 + 16 + l16) * DD + dk + koffB);
            s0 = __builtin_amdgcn_wmma_f32_16x16x32_bf16(false, a, false, b0,
                                                         (short)0, s0, false, false);
            s1 = __builtin_amdgcn_wmma_f32_16x16x32_bf16(false, a, false, b1,
                                                         (short)0, s1, false, false);
        }
        // ---- scale + causal mask (only diagonal-straddling chunk)
        if (k0 + 31 > q0 + 15) {
#pragma unroll
            for (int r = 0; r < 8; ++r) {
                int qrow = q0 + r + 8 * half;
                s0[r] = (k0 + l16      <= qrow) ? s0[r] * sc : -__builtin_inff();
                s1[r] = (k0 + 16 + l16 <= qrow) ? s1[r] * sc : -__builtin_inff();
            }
        } else {
#pragma unroll
            for (int r = 0; r < 8; ++r) { s0[r] *= sc; s1[r] *= sc; }
        }
        // ---- online softmax; C-tile rows live in 16-lane halves
#pragma unroll
        for (int r = 0; r < 8; ++r) {
            float cm = fmaxf(s0[r], s1[r]);
#pragma unroll
            for (int off = 1; off < 16; off <<= 1)
                cm = fmaxf(cm, __shfl_xor(cm, off, 32));
            float mn    = fmaxf(rowmax[r], cm);
            float alpha = __expf(rowmax[r] - mn);
            float p0    = __expf(s0[r] - mn);
            float p1    = __expf(s1[r] - mn);
            float ps    = p0 + p1;
#pragma unroll
            for (int off = 1; off < 16; off <<= 1)
                ps += __shfl_xor(ps, off, 32);
            rowsum[r] = rowsum[r] * alpha + ps;
            rowmax[r] = mn;
#pragma unroll
            for (int t = 0; t < 8; ++t) acc[t][r] *= alpha;
            int row = r + 8 * half;
            lds[wave][row][l16]      = f2bf(p0);
            lds[wave][row][l16 + 16] = f2bf(p1);
        }
        // DS ops are in-order within a wave; fence keeps the compiler honest
        asm volatile("s_wait_dscnt 0x0" ::: "memory");
        v8bf plo = *(const v8bf*)(&lds[wave][l16][koffA]);
        v8bf phi = *(const v8bf*)(&lds[wave][l16][koffA + 8]);
        v16bf pa;
#pragma unroll
        for (int i = 0; i < 8; ++i) { pa[i] = plo[i]; pa[8 + i] = phi[i]; }
        // ---- O += P.V
#pragma unroll
        for (int t = 0; t < 8; ++t) {
            v16bf bv = *(const v16bf*)(vbase + (size_t)(t * 16 + l16) * SS + k0 + koffB);
            acc[t] = __builtin_amdgcn_wmma_f32_16x16x32_bf16(false, pa, false, bv,
                                                             (short)0, acc[t], false, false);
        }
    }

    // ---- normalize and store context (row-major feeds the O-proj GEMM)
    const size_t obase = (size_t)(b * SS + q0) * DD + h * HD;
#pragma unroll
    for (int r = 0; r < 8; ++r) {
        float inv = 1.0f / rowsum[r];
        int row = r + 8 * half;
#pragma unroll
        for (int t = 0; t < 8; ++t)
            O[obase + (size_t)row * DD + t * 16 + l16] = f2bf(acc[t][r] * inv);
    }
}

// ---------------------------------------------------------------------------

extern "C" void kernel_launch(void* const* d_in, const int* in_sizes, int n_in,
                              void* d_out, int out_size, void* d_ws, size_t ws_size,
                              hipStream_t stream) {
    const float* x    = (const float*)d_in[0];
    const float* cosT = (const float*)d_in[1];
    const float* sinT = (const float*)d_in[2];
    const float* Wq   = (const float*)d_in[3];
    const float* bq   = (const float*)d_in[4];
    const float* Wk   = (const float*)d_in[5];
    const float* bk   = (const float*)d_in[6];
    const float* Wv   = (const float*)d_in[7];
    const float* bv   = (const float*)d_in[8];
    const float* Wo   = (const float*)d_in[9];
    const float* bo   = (const float*)d_in[10];
    float* out = (float*)d_out;

    char* ws = (char*)d_ws;
    size_t off = 0;
    auto alloc = [&](size_t bytes) {
        void* p = ws + off;
        off += (bytes + 255) & ~(size_t)255;
        return p;
    };
    __bf16* Wqt = (__bf16*)alloc((size_t)DD * DD * 2);
    __bf16* Wkt = (__bf16*)alloc((size_t)DD * DD * 2);
    __bf16* Wvt = (__bf16*)alloc((size_t)DD * DD * 2);
    __bf16* Wot = (__bf16*)alloc((size_t)DD * DD * 2);
    __bf16* xb  = (__bf16*)alloc((size_t)BS * DD * 2);
    float*  Qf  = (float*) alloc((size_t)BS * DD * 4);
    float*  Kf  = (float*) alloc((size_t)BS * DD * 4);
    __bf16* Vt  = (__bf16*)alloc((size_t)BS * DD * 2);
    __bf16* Qb  = (__bf16*)alloc((size_t)BS * DD * 2);
    __bf16* Kb  = (__bf16*)alloc((size_t)BS * DD * 2);
    __bf16* Cb  = (__bf16*)alloc((size_t)BS * DD * 2);

    const int wBlocks = (DD * DD) / 256;             // 16384
    const int xBlocks = (BS * DD) / 256;             // 32768
    const dim3 gGrid(BS / 64, DD / 128);             // 64 x 16 blocks
    const int aBlocks = (BB * HH * (SS / 16)) / 8;   // 512

    // 1) precision conversion
    cvt_wt_kernel<<<wBlocks, 256, 0, stream>>>(Wq, Wqt);
    cvt_wt_kernel<<<wBlocks, 256, 0, stream>>>(Wk, Wkt);
    cvt_wt_kernel<<<wBlocks, 256, 0, stream>>>(Wv, Wvt);
    cvt_wt_kernel<<<wBlocks, 256, 0, stream>>>(Wo, Wot);
    cvt_bf16_kernel<<<xBlocks, 256, 0, stream>>>(x, xb);

    // 2) QKV projections (WMMA + double-buffered async LDS staging)
    gemm_kernel<<<gGrid, 256, 0, stream>>>(xb, Wqt, bq, Qf, nullptr, 0);
    gemm_kernel<<<gGrid, 256, 0, stream>>>(xb, Wkt, bk, Kf, nullptr, 0);
    gemm_kernel<<<gGrid, 256, 0, stream>>>(xb, Wvt, bv, nullptr, Vt, 1);

    // 3) RoPE -> bf16 Q/K
    rope_kernel<<<xBlocks, 256, 0, stream>>>(Qf, cosT, sinT, Qb);
    rope_kernel<<<xBlocks, 256, 0, stream>>>(Kf, cosT, sinT, Kb);

    // 4) causal flash attention (WMMA)
    fattn_kernel<<<aBlocks, 256, 0, stream>>>(Qb, Kb, Vt, Cb);

    // 5) output projection (WMMA) -> fp32 result
    gemm_kernel<<<gGrid, 256, 0, stream>>>(Cb, Wot, bo, out, nullptr, 0);
}
